// EHRIntervalEncoder_11622181503398
// MI455X (gfx1250) — matrix operand order repaired and checked by reading, not compiled, not run on hardware
//
#include <hip/hip_runtime.h>

// ---------------- Shapes ----------------
#define BB 4
#define LL 256
#define DD 256
#define PP 32
#define HH 8
#define HD 32
#define DH 128          // D/2

typedef __attribute__((ext_vector_type(16))) _Float16 v16h;
typedef __attribute__((ext_vector_type(8)))  float    v8f;

__device__ __forceinline__ v16h load16h(const _Float16* p) {
  // 16 contiguous halfs (32B)
  union { uint4 u[2]; v16h v; } x;
  x.u[0] = *(const uint4*)(p);
  x.u[1] = *(const uint4*)(p + 8);
  return x.v;
}
__device__ __forceinline__ v16h load2x8h(const _Float16* p0, const _Float16* p1) {
  // two contiguous 8-half (16B) chunks
  union { uint4 u[2]; v16h v; } x;
  x.u[0] = *(const uint4*)(p0);
  x.u[1] = *(const uint4*)(p1);
  return x.v;
}
__device__ __forceinline__ v8f wmma16(v16h a, v16h b, v8f c) {
  return __builtin_amdgcn_wmma_f32_16x16x32_f16(false, a, false, b, (short)0, c, false, false);
}
__device__ __forceinline__ void lds_fence() { asm volatile("" ::: "memory"); }

// ---------------- Prep kernels ----------------
__global__ void __launch_bounds__(256) cvt_f16_kernel(const float* __restrict__ in,
                                                      _Float16* __restrict__ out, int n) {
  int i = blockIdx.x * 256 + threadIdx.x;
  if (i < n) out[i] = (_Float16)in[i];
}

// in: R x C row-major fp32 ; out: C x R row-major f16 (transposed, scaled)
__global__ void __launch_bounds__(256) transpose_f16_kernel(const float* __restrict__ in,
                                                            _Float16* __restrict__ out,
                                                            int R, int C, float scale) {
  int i = blockIdx.x * 256 + threadIdx.x;
  if (i < R * C) {
    int r = i / C, c = i % C;
    out[(size_t)c * R + r] = (_Float16)(in[i] * scale);
  }
}

// ---------------- Generic f16 WMMA GEMM:  C = A(MxK) * BT(NxK)^T + bias  ----------------
// N == 256, K == 256. grid.x = M/16, block = 256 (8 waves), each wave 2 N-tiles.
template <bool OUT_F32>
__global__ void __launch_bounds__(256) gemm16_f16_kernel(
    const _Float16* __restrict__ A, const _Float16* __restrict__ BT,
    const float* __restrict__ bias, float bias_scale,
    void* __restrict__ Cout, int M) {
  const int K = 256, N = 256;
  __shared__ _Float16 Alds[16 * 264];
  const int m0 = blockIdx.x * 16;
  const int t = threadIdx.x;
  { // stage A tile (16 x 256) into LDS with gfx1250 async memory->LDS DMA
    const int row = t >> 4, c0 = (t & 15) * 16;
    const unsigned goff = (unsigned)(((size_t)(m0 + row) * K + c0) * sizeof(_Float16));
    const unsigned ldso = (unsigned)(size_t)(&Alds[row * 264 + c0]);
    asm volatile(
        "global_load_async_to_lds_b128 %0, %1, %2\n\t"
        "global_load_async_to_lds_b128 %0, %1, %2 offset:16"
        :: "v"(ldso), "v"(goff), "s"(A) : "memory");
    asm volatile("s_wait_asynccnt 0x0" ::: "memory");
  }
  __syncthreads();
  const int lane = t & 31, wave = t >> 5;
  const int ln = lane & 15, hi = lane >> 4;
  // preload all 8 A fragments once; reuse for both N-tiles
  v16h afrag[8];
#pragma unroll
  for (int kk = 0; kk < 8; ++kk)
    afrag[kk] = load2x8h(&Alds[ln * 264 + kk * 32 + hi * 16],
                         &Alds[ln * 264 + kk * 32 + hi * 16 + 8]);
  const int n1 = (wave * 2 + 1) * 16;
  __builtin_prefetch(&BT[(size_t)(n1 + ln) * K], 0, 0);  // global_prefetch_b8
  for (int i = 0; i < 2; ++i) {
    const int n0 = (wave * 2 + i) * 16;
    v8f c = {};
#pragma unroll
    for (int kk = 0; kk < 8; ++kk) {
      v16h b = load16h(&BT[(size_t)(n0 + ln) * K + kk * 32 + hi * 16]);
      c = wmma16(afrag[kk], b, c);
    }
    const float bv = bias[n0 + ln] * bias_scale;
    if (OUT_F32) {
      float* C = (float*)Cout;
#pragma unroll
      for (int j = 0; j < 8; ++j)
        C[(size_t)(m0 + j + hi * 8) * N + n0 + ln] = c[j] + bv;
    } else {
      _Float16* C = (_Float16*)Cout;
#pragma unroll
      for (int j = 0; j < 8; ++j)
        C[(size_t)(m0 + j + hi * 8) * N + n0 + ln] = (_Float16)(c[j] + bv);
    }
  }
}

// ---------------- Time-feature MLP + interval bias ----------------
// grid = B*P*2 (half of L per block), block = 128 (4 waves)
__global__ void __launch_bounds__(128) mlp_time_kernel(
    const float* __restrict__ ehr_times, const float* __restrict__ interval_times,
    const float* __restrict__ w1, const float* __restrict__ b1,
    const _Float16* __restrict__ w2T, const float* __restrict__ b2,
    _Float16* __restrict__ temb, float* __restrict__ biasbuf) {
  __shared__ _Float16 Hlds[128 * 136];
  const int blk = blockIdx.x;
  const int half = blk & 1;
  const int bp = blk >> 1;          // b*P + p
  const int b = bp >> 5;            // P = 32
  const int l0 = half * 128;
  const int t = threadIdx.x;
  const float s = interval_times[bp * 2 + 0];
  const float e = interval_times[bp * 2 + 1];
  {
    const int l = l0 + t;
    const float et = ehr_times[b * LL + l];
    const float ds = et - s, de = e - et;
    const float sg = 1.0f / (1.0f + __expf(-(ds * de)));
#pragma unroll 4
    for (int j = 0; j < DH; ++j) {
      float pre = ds * w1[j] + de * w1[DH + j] + sg * w1[2 * DH + j] + b1[j];
      float g = 0.5f * pre * (1.0f + erff(pre * 0.70710678118654752f)); // exact GELU
      Hlds[t * 136 + j] = (_Float16)g;
    }
    const float c = 0.5f * (s + e);
    biasbuf[(size_t)bp * LL + l] =
        (et >= s && et <= e) ? -fabsf(et - c) : -__builtin_inff();
  }
  __syncthreads();
  const int lane = t & 31, wave = t >> 5;
  const int ln = lane & 15, hi = lane >> 4;
  _Float16* outbase = temb + ((size_t)bp * LL + l0) * DD;
  for (int tile = wave; tile < 128; tile += 4) {   // 8 m-tiles x 16 n-tiles
    const int m0 = (tile >> 4) * 16, n0 = (tile & 15) * 16;
    v8f c = {};
#pragma unroll
    for (int k0 = 0; k0 < DH; k0 += 32) {
      v16h a = load2x8h(&Hlds[(m0 + ln) * 136 + k0 + hi * 16],
                        &Hlds[(m0 + ln) * 136 + k0 + hi * 16 + 8]);
      v16h b = load16h(&w2T[(size_t)(n0 + ln) * DH + k0 + hi * 16]);
      c = wmma16(a, b, c);
    }
    const float bv = b2[n0 + ln];
#pragma unroll
    for (int j = 0; j < 8; ++j)
      outbase[(size_t)(m0 + j + hi * 8) * DD + n0 + ln] = (_Float16)(c[j] + bv);
  }
}

// ---------------- Attention per (b,p,h) ----------------
#define SSTR 260   // score row stride (floats)
#define VSTR 264   // vT row stride (halfs)
__global__ void __launch_bounds__(64) attn_kernel(
    const _Float16* __restrict__ qbuf, const _Float16* __restrict__ kbuf,
    const _Float16* __restrict__ vbuf, const float* __restrict__ biasbuf,
    _Float16* __restrict__ ctx) {
  __shared__ _Float16 vT[32 * VSTR];        // v^T: [d][key]
  __shared__ float scores[2 * 16 * SSTR];   // per-wave 16 x 256 score rows
  const int blk = blockIdx.x;               // (b*P + p)*H + h
  const int h = blk & (HH - 1);
  const int bp = blk >> 3;
  const int b = bp >> 5;
  const int t = threadIdx.x, lane = t & 31, wave = t >> 5;
  const int ln = lane & 15, hi = lane >> 4;
  const _Float16* qb = qbuf + (size_t)bp * LL * DD + h * HD;
  const _Float16* kb = kbuf + (size_t)b * LL * DD + h * HD;
  const _Float16* vb = vbuf + (size_t)b * LL * DD + h * HD;
  const float* br = biasbuf + (size_t)bp * LL;
  // stage v^T into LDS
  for (int key = t; key < LL; key += 64) {
    union { uint4 u[4]; _Float16 x[32]; } vr;
    const uint4* src = (const uint4*)(vb + (size_t)key * DD);
    vr.u[0] = src[0]; vr.u[1] = src[1]; vr.u[2] = src[2]; vr.u[3] = src[3];
#pragma unroll
    for (int d = 0; d < HD; ++d) vT[d * VSTR + key] = vr.x[d];
  }
  __syncthreads();

  float* sw = scores + wave * 16 * SSTR;
  _Float16* aw = (_Float16*)sw;             // attn alias, row stride 2*SSTR halfs
  for (int rt = wave; rt < 16; rt += 2) {
    const int m0 = rt * 16;
    // A fragment of q (16 rows x K=32), reused for all 16 key tiles
    v16h aq = load2x8h(qb + (size_t)(m0 + ln) * DD + hi * 16,
                       qb + (size_t)(m0 + ln) * DD + hi * 16 + 8);
#pragma unroll 1
    for (int nt = 0; nt < 16; ++nt) {
      const int n0 = nt * 16;
      v16h bk = load16h(kb + (size_t)(n0 + ln) * DD + hi * 16);
      v8f c = {};
      c = wmma16(aq, bk, c);
      const float bv = br[n0 + ln];
#pragma unroll
      for (int j = 0; j < 8; ++j)
        sw[(j + hi * 8) * SSTR + n0 + ln] = c[j] + bv;
    }
    lds_fence();
    // fp32 softmax per row; write f16 attn aliased over the score rows
#pragma unroll 1
    for (int r = 0; r < 16; ++r) {
      float* row = sw + r * SSTR;
      float x0[8], ex[8];
      float mx = -__builtin_inff();
#pragma unroll
      for (int i = 0; i < 8; ++i) { x0[i] = row[lane * 8 + i]; mx = fmaxf(mx, x0[i]); }
#pragma unroll
      for (int off = 16; off > 0; off >>= 1) mx = fmaxf(mx, __shfl_xor(mx, off, 32));
      float sum = 0.0f;
      if (mx == -__builtin_inff()) {
#pragma unroll
        for (int i = 0; i < 8; ++i) ex[i] = 0.0f;
        sum = 1.0f;                                  // row fully masked -> zeros
      } else {
#pragma unroll
        for (int i = 0; i < 8; ++i) { ex[i] = __expf(x0[i] - mx); sum += ex[i]; }
      }
#pragma unroll
      for (int off = 16; off > 0; off >>= 1) sum += __shfl_xor(sum, off, 32);
      const float rn = 1.0f / sum;
      _Float16* arow = aw + r * (2 * SSTR);
#pragma unroll
      for (int i = 0; i < 8; ++i) arow[lane * 8 + i] = (_Float16)(ex[i] * rn);
    }
    lds_fence();
    // ctx = attn (16 x 256) @ v (256 x 32)
#pragma unroll 1
    for (int d0 = 0; d0 < HD; d0 += 16) {
      v8f c = {};
#pragma unroll
      for (int k0 = 0; k0 < LL; k0 += 32) {
        v16h a = load2x8h(aw + ln * (2 * SSTR) + k0 + hi * 16,
                          aw + ln * (2 * SSTR) + k0 + hi * 16 + 8);
        v16h bv = load16h(&vT[(d0 + ln) * VSTR + k0 + hi * 16]);
        c = wmma16(a, bv, c);
      }
#pragma unroll
      for (int j = 0; j < 8; ++j)
        ctx[((size_t)bp * LL + m0 + j + hi * 8) * DD + h * HD + d0 + ln] = (_Float16)c[j];
    }
    lds_fence();
  }
}

// ---------------- Host launcher ----------------
extern "C" void kernel_launch(void* const* d_in, const int* in_sizes, int n_in,
                              void* d_out, int out_size, void* d_ws, size_t ws_size,
                              hipStream_t stream) {
  const float* ehr   = (const float*)d_in[0];   // (B,L,D)
  const float* ehr_t = (const float*)d_in[1];   // (B,L)
  const float* itv   = (const float*)d_in[2];   // (B,P,2)
  const float* w1    = (const float*)d_in[3];   // (3,128)
  const float* b1    = (const float*)d_in[4];
  const float* w2    = (const float*)d_in[5];   // (128,256)
  const float* b2    = (const float*)d_in[6];
  const float* wq    = (const float*)d_in[7];
  const float* bq    = (const float*)d_in[8];
  const float* wk    = (const float*)d_in[9];
  const float* bk    = (const float*)d_in[10];
  const float* wv    = (const float*)d_in[11];
  const float* bv    = (const float*)d_in[12];
  const float* wo    = (const float*)d_in[13];
  const float* bo    = (const float*)d_in[14];
  float* out = (float*)d_out;

  const float scale = 0.17677669529663687f;  // HD^-0.5

  // workspace carve-up (256B aligned)
  size_t off = 0;
  char* ws = (char*)d_ws;
  auto carve = [&](size_t bytes) -> char* {
    char* p = ws + off;
    off += (bytes + 255) & ~(size_t)255;
    return p;
  };
  _Float16* ehr16 = (_Float16*)carve((size_t)BB * LL * DD * 2);      // 512KB
  _Float16* wkT   = (_Float16*)carve((size_t)DD * DD * 2);
  _Float16* wvT   = (_Float16*)carve((size_t)DD * DD * 2);
  _Float16* wqT   = (_Float16*)carve((size_t)DD * DD * 2);
  _Float16* woT   = (_Float16*)carve((size_t)DD * DD * 2);
  _Float16* w2T   = (_Float16*)carve((size_t)DD * DH * 2);
  _Float16* kbuf  = (_Float16*)carve((size_t)BB * LL * DD * 2);
  _Float16* vbuf  = (_Float16*)carve((size_t)BB * LL * DD * 2);
  float*    biasb = (float*)   carve((size_t)BB * PP * LL * 4);
  _Float16* temb  = (_Float16*)carve((size_t)BB * PP * LL * DD * 2); // 16MB
  _Float16* qbuf  = (_Float16*)carve((size_t)BB * PP * LL * DD * 2); // 16MB
  _Float16* ctxb  = temb;  // ctx reuses time_emb storage (dead after q proj)
  if (off > ws_size) return;

  // prep: conversions / transposes
  cvt_f16_kernel<<<(BB * LL * DD) / 256, 256, 0, stream>>>(ehr, ehr16, BB * LL * DD);
  transpose_f16_kernel<<<(DD * DD) / 256, 256, 0, stream>>>(wk, wkT, DD, DD, 1.0f);
  transpose_f16_kernel<<<(DD * DD) / 256, 256, 0, stream>>>(wv, wvT, DD, DD, 1.0f);
  transpose_f16_kernel<<<(DD * DD) / 256, 256, 0, stream>>>(wq, wqT, DD, DD, scale);
  transpose_f16_kernel<<<(DD * DD) / 256, 256, 0, stream>>>(wo, woT, DD, DD, 1.0f);
  transpose_f16_kernel<<<(DH * DD) / 256, 256, 0, stream>>>(w2, w2T, DH, DD, 1.0f);

  // k, v projections: (B*L=1024) x 256 @ 256 x 256
  gemm16_f16_kernel<false><<<(BB * LL) / 16, 256, 0, stream>>>(ehr16, wkT, bk, 1.0f, kbuf, BB * LL);
  gemm16_f16_kernel<false><<<(BB * LL) / 16, 256, 0, stream>>>(ehr16, wvT, bv, 1.0f, vbuf, BB * LL);

  // time-feature MLP -> time_emb (f16) + interval bias (f32)
  mlp_time_kernel<<<BB * PP * 2, 128, 0, stream>>>(ehr_t, itv, w1, b1, w2T, b2, temb, biasb);

  // q projection (scale folded): (B*P*L=32768) x 256 @ 256 x 256
  gemm16_f16_kernel<false><<<(BB * PP * LL) / 16, 256, 0, stream>>>(temb, wqT, bq, scale, qbuf, BB * PP * LL);

  // attention per (b,p,h)
  attn_kernel<<<BB * PP * HH, 64, 0, stream>>>(qbuf, kbuf, vbuf, biasb, ctxb);

  // output projection -> fp32 d_out
  gemm16_f16_kernel<true><<<(BB * PP * LL) / 16, 256, 0, stream>>>(ctxb, woT, bo, 1.0f, out, BB * PP * LL);
}
